// FP8Linear_36996848287756
// MI455X (gfx1250) — 8-bit matrix operand order repaired and checked by reading, compiled, not run on hardware
//
#include <hip/hip_runtime.h>


typedef __attribute__((ext_vector_type(8))) int v8i;
typedef int vs4i __attribute__((vector_size(16)));

#define TILE_M 128
#define TILE_N 128
#define TILE_K 64
#define LDS_STRIDE 80

#define AS1 __attribute__((address_space(1)))
#define AS3 __attribute__((address_space(3)))

#if defined(__has_builtin)
#  if __has_builtin(__builtin_amdgcn_global_load_async_to_lds_b128)
#    define USE_ASYNC_LDS 1
#  endif
#endif
#ifndef USE_ASYNC_LDS
#  define USE_ASYNC_LDS 0
#endif

__device__ __forceinline__ void copy16_g2l(signed char* lds, const signed char* g) {
#if USE_ASYNC_LDS
    __builtin_amdgcn_global_load_async_to_lds_b128(
        (AS1 vs4i*)g, (AS3 vs4i*)lds,  0,  0);
#else
    *(int4*)lds = *(const int4*)g;
#endif
}

__device__ __forceinline__ void wait_async_all() {
#if USE_ASYNC_LDS
#  if __has_builtin(__builtin_amdgcn_s_wait_asynccnt)
    __builtin_amdgcn_s_wait_asynccnt(0);
#  else
    asm volatile("s_wait_asynccnt 0" ::: "memory");
#  endif
#endif
}

__global__ void init_scales_kernel(unsigned* s) {
    if (threadIdx.x < 2) s[threadIdx.x] = 0u;
}

__global__ void amax_kernel(const float4* __restrict__ src, unsigned* __restrict__ dst, int n4) {
    __shared__ float red[256];
    float m = 0.0f;
    for (int i = blockIdx.x * blockDim.x + threadIdx.x; i < n4; i += gridDim.x * blockDim.x) {
        float4 v = src[i];
        m = fmaxf(m, fmaxf(fmaxf(fabsf(v.x), fabsf(v.y)), fmaxf(fabsf(v.z), fabsf(v.w))));
    }
    red[threadIdx.x] = m;
    __syncthreads();
    for (int s = 128; s > 0; s >>= 1) {
        if (threadIdx.x < s) red[threadIdx.x] = fmaxf(red[threadIdx.x], red[threadIdx.x + s]);
        __syncthreads();
    }
    if (threadIdx.x == 0) atomicMax(dst, __float_as_uint(red[0]));
}

__global__ void quant_kernel(const float4* __restrict__ src, int* __restrict__ dst,
                             const unsigned* __restrict__ amax_bits, int n4) {
    const float amax = fmaxf(__uint_as_float(amax_bits[0]), 1e-12f);
    const float r = 127.0f / amax;
    for (int i = blockIdx.x * blockDim.x + threadIdx.x; i < n4; i += gridDim.x * blockDim.x) {
        float4 v = src[i];
        int q0 = (int)rintf(fminf(fmaxf(v.x * r, -127.0f), 127.0f));
        int q1 = (int)rintf(fminf(fmaxf(v.y * r, -127.0f), 127.0f));
        int q2 = (int)rintf(fminf(fmaxf(v.z * r, -127.0f), 127.0f));
        int q3 = (int)rintf(fminf(fmaxf(v.w * r, -127.0f), 127.0f));
        dst[i] = (q0 & 255) | ((q1 & 255) << 8) | ((q2 & 255) << 16) | ((q3 & 255) << 24);
    }
}

__global__ void __launch_bounds__(256)
gemm_i8_wmma_kernel(const signed char* __restrict__ Aq,
                    const signed char* __restrict__ Wq,
                    const unsigned* __restrict__ amax_bits,
                    float* __restrict__ out,
                    int K, int N) {
    __shared__ signed char smA[2][TILE_M * LDS_STRIDE];
    __shared__ signed char smB[2][TILE_N * LDS_STRIDE];

    const int tid     = threadIdx.x;
    const int lane    = tid & 31;
    const int wave    = tid >> 5;
    const int waveM   = wave >> 1;
    const int waveN   = wave & 1;
    const int laneM   = lane & 15;
    const int halfSel = lane >> 4;

    const long blockRow = (long)blockIdx.y * TILE_M;
    const long blockCol = (long)blockIdx.x * TILE_N;

    const int ldRow = tid >> 2;
    const int ldOff = (tid & 3) << 4;

    v8i acc[2][4];
    const v8i zero = {0, 0, 0, 0, 0, 0, 0, 0};
#pragma unroll
    for (int i = 0; i < 2; ++i)
#pragma unroll
        for (int j = 0; j < 4; ++j) acc[i][j] = zero;

    const int kSteps = K / TILE_K;

    auto issue_tile = [&](int kt, int buf) {
        const long kBase = (long)kt * TILE_K;
#pragma unroll
        for (int h = 0; h < 2; ++h) {
            const int r = ldRow + h * 64;
            copy16_g2l(&smA[buf][r * LDS_STRIDE + ldOff],
                       Aq + (blockRow + r) * (long)K + kBase + ldOff);
            copy16_g2l(&smB[buf][r * LDS_STRIDE + ldOff],
                       Wq + (blockCol + r) * (long)K + kBase + ldOff);
        }
    };

    issue_tile(0, 0);
    wait_async_all();
    __syncthreads();

    for (int kt = 0; kt < kSteps; ++kt) {
        const int buf = kt & 1;
        if (kt + 1 < kSteps) issue_tile(kt + 1, buf ^ 1);

        v8i afrag[2];
#pragma unroll
        for (int i = 0; i < 2; ++i) {
            const signed char* base =
                &smA[buf][(waveM * 32 + i * 16 + laneM) * LDS_STRIDE] + (halfSel << 3);
#pragma unroll
            for (int v = 0; v < 8; ++v) {
                const int kOff = ((v >> 1) << 4) + ((v & 1) << 2);
                afrag[i][v] = *(const int*)(base + kOff);
            }
        }
        v8i bfrag[4];
#pragma unroll
        for (int j = 0; j < 4; ++j) {
            const signed char* base =
                &smB[buf][(waveN * 64 + j * 16 + laneM) * LDS_STRIDE] + (halfSel << 3);
#pragma unroll
            for (int v = 0; v < 8; ++v) {
                const int kOff = ((v >> 1) << 4) + ((v & 1) << 2);
                bfrag[j][v] = *(const int*)(base + kOff);
            }
        }

#pragma unroll
        for (int i = 0; i < 2; ++i)
#pragma unroll
            for (int j = 0; j < 4; ++j)
                acc[i][j] = __builtin_amdgcn_wmma_i32_16x16x64_iu8(
                     true, afrag[i],  true, bfrag[j],
                    acc[i][j],  false,  false);

        wait_async_all();
        __syncthreads();
    }

    const float factor = fmaxf(__uint_as_float(amax_bits[0]), 1e-12f) *
                         fmaxf(__uint_as_float(amax_bits[1]), 1e-12f) * (1.0f / 16129.0f);

#pragma unroll
    for (int i = 0; i < 2; ++i) {
#pragma unroll
        for (int j = 0; j < 4; ++j) {
            const long rBase = blockRow + waveM * 32 + i * 16 + halfSel * 8;
            const long c     = blockCol + waveN * 64 + j * 16 + laneM;
#pragma unroll
            for (int e = 0; e < 8; ++e)
                out[(rBase + e) * (long)N + c] = (float)acc[i][j][e] * factor;
        }
    }
}

extern "C" void kernel_launch(void* const* d_in, const int* in_sizes, int n_in,
                              void* d_out, int out_size, void* d_ws, size_t ws_size,
                              hipStream_t stream) {
    const float* x = (const float*)d_in[0];
    const float* w = (const float*)d_in[1];
    float* out = (float*)d_out;

    const int  K = 4096;
    const long xN = (long)in_sizes[0];
    const long wN = (long)in_sizes[1];
    const int  M = (int)(xN / K);
    const int  N = (int)(wN / K);

    unsigned*    amaxs = (unsigned*)d_ws;
    signed char* Aq    = (signed char*)d_ws + 256;
    signed char* Wq    = Aq + (size_t)M * (size_t)K;

    init_scales_kernel<<<1, 32, 0, stream>>>(amaxs);
    amax_kernel<<<2048, 256, 0, stream>>>((const float4*)x, amaxs + 0, (int)(xN / 4));
    amax_kernel<<<2048, 256, 0, stream>>>((const float4*)w, amaxs + 1, (int)(wN / 4));
    quant_kernel<<<4096, 256, 0, stream>>>((const float4*)x, (int*)Aq, amaxs + 0, (int)(xN / 4));
    quant_kernel<<<4096, 256, 0, stream>>>((const float4*)w, (int*)Wq, amaxs + 1, (int)(wN / 4));

    dim3 grid(N / TILE_N, M / TILE_M);
    gemm_i8_wmma_kernel<<<grid, 256, 0, stream>>>(Aq, Wq, amaxs, out, K, N);
}
